// SRUPosDouble_68066641707709
// MI455X (gfx1250) — compile-verified, matrix operand order checked by reading, not run on hardware
//
#include <hip/hip_runtime.h>
#include <hip/hip_bf16.h>

// Problem constants (from reference)
#define TB   512            // T (recurrence length; batch dim acts as time)
#define NN   256            // N (parallel dim inside each layer)
#define ROWS (TB * NN)      // 131072 rows for every GEMM

typedef __bf16 bf16_t;
typedef __attribute__((ext_vector_type(8)))  bf16_t bf16x8;
typedef __attribute__((ext_vector_type(16))) bf16_t afrag16;   // WMMA A/B fragment
typedef __attribute__((ext_vector_type(8)))  float  accf8;     // WMMA C/D fragment

// ---------------- helpers ----------------

__device__ __forceinline__ unsigned short f2bf_rne(float x) {
  unsigned int u = __float_as_uint(x);
  u += 0x7FFFu + ((u >> 16) & 1u);      // round-to-nearest-even
  return (unsigned short)(u >> 16);
}

__device__ __forceinline__ float wave_sum(float v) {
#pragma unroll
  for (int o = 16; o > 0; o >>= 1) v += __shfl_xor(v, o, 32);
  return v;
}

// Load one 16-bit WMMA A/B fragment for row `rowPtr` at K-offset k0.
// Per ISA layout: lane half L holds K = k0 + L*8 .. +7 (elems 0..7) and
// K = k0 + 16 + L*8 .. +7 (elems 8..15), two contiguous 16B loads.
__device__ __forceinline__ afrag16 load_frag(const unsigned short* rowPtr, int k0, int half) {
  bf16x8 lo = *reinterpret_cast<const bf16x8*>(rowPtr + k0 + half * 8);
  bf16x8 hi = *reinterpret_cast<const bf16x8*>(rowPtr + k0 + 16 + half * 8);
  return __builtin_shufflevector(lo, hi, 0, 1, 2, 3, 4, 5, 6, 7,
                                         8, 9, 10, 11, 12, 13, 14, 15);
}

// ---------------- kernels ----------------

// Embedding gather + LayerNorm (D = 256). One wave per row, 8 waves/block.
__global__ void embed_ln_kernel(const int* __restrict__ x,
                                const float* __restrict__ emb,
                                const float* __restrict__ g,
                                const float* __restrict__ b,
                                float* __restrict__ out) {
  const int D = 256;
  int row  = blockIdx.x * 8 + (threadIdx.x >> 5);
  int lane = threadIdx.x & 31;
  const float* src = emb + (long)x[row] * D;
  float vals[8];
  float s = 0.f;
#pragma unroll
  for (int i = 0; i < 8; i++) { vals[i] = src[lane + i * 32]; s += vals[i]; }
  s = wave_sum(s);
  float mean = s * (1.0f / D);
  float q = 0.f;
#pragma unroll
  for (int i = 0; i < 8; i++) { float d = vals[i] - mean; q += d * d; }
  q = wave_sum(q);
  float rstd = rsqrtf(q * (1.0f / D) + 1e-5f);
  float* dst = out + (long)row * D;
#pragma unroll
  for (int i = 0; i < 8; i++) {
    int d = lane + i * 32;
    dst[d] = (vals[i] - mean) * rstd * g[d] + b[d];
  }
}

// LayerNorm producing both f32 (for SRU residual) and bf16 (for WMMA A).
template <int D>
__global__ void ln_kernel(const float* __restrict__ in,
                          const float* __restrict__ g,
                          const float* __restrict__ beta,
                          float* __restrict__ xn,
                          unsigned short* __restrict__ xnb) {
  constexpr int PER = D / 32;
  int row  = blockIdx.x * 8 + (threadIdx.x >> 5);
  int lane = threadIdx.x & 31;
  const float* src = in + (long)row * D;
  float vals[PER];
  float s = 0.f;
#pragma unroll
  for (int i = 0; i < PER; i++) { vals[i] = src[lane + i * 32]; s += vals[i]; }
  s = wave_sum(s);
  float mean = s * (1.0f / D);
  float q = 0.f;
#pragma unroll
  for (int i = 0; i < PER; i++) { float d = vals[i] - mean; q += d * d; }
  q = wave_sum(q);
  float rstd = rsqrtf(q * (1.0f / D) + 1e-5f);
  float* dx = xn + (long)row * D;
  unsigned short* db = xnb + (long)row * D;
#pragma unroll
  for (int i = 0; i < PER; i++) {
    int d = lane + i * 32;
    float v = (vals[i] - mean) * rstd * g[d] + beta[d];
    dx[d] = v;
    db[d] = f2bf_rne(v);
  }
}

// Transpose + convert weight: W (K x Ncols, f32, K-major rows) -> Wt[col][k] bf16
__global__ void wconv_kernel(const float* __restrict__ W,
                             unsigned short* __restrict__ Wt,
                             int K, int Ncols) {
  int idx = blockIdx.x * 256 + threadIdx.x;
  if (idx >= K * Ncols) return;
  int c = idx / K, d = idx % K;
  Wt[idx] = f2bf_rne(W[(long)d * Ncols + c]);
}

// Flat f32 -> bf16 convert
__global__ void f2b_kernel(const float* __restrict__ in,
                           unsigned short* __restrict__ out, long n) {
  long idx = (long)blockIdx.x * 256 + threadIdx.x;
  if (idx < n) out[idx] = f2bf_rne(in[idx]);
}

// Tiled bf16 WMMA GEMM: C[M x NCOLS] = A[M x K] * Wt[NCOLS x K]^T, f32 accum.
// Block = 256 threads = 8 waves; each wave owns a 16(M) x 16*NTILES(N) strip,
// so one A fragment feeds NTILES WMMAs per K-step.
template <int K, int NCOLS, int NTILES>
__global__ void gemm_bf16_kernel(const unsigned short* __restrict__ A,
                                 const unsigned short* __restrict__ Wt,
                                 float* __restrict__ C) {
  int wave = threadIdx.x >> 5, lane = threadIdx.x & 31;
  int half = lane >> 4, l15 = lane & 15;
  int mBase = blockIdx.x * 128 + wave * 16;
  int nBase = blockIdx.y * (16 * NTILES);
  const unsigned short* aRow  = A  + (long)(mBase + l15) * K;
  const unsigned short* bRow0 = Wt + (long)(nBase + l15) * K;
  accf8 acc[NTILES] = {};
#pragma unroll
  for (int k0 = 0; k0 < K; k0 += 32) {
    afrag16 a = load_frag(aRow, k0, half);
#pragma unroll
    for (int i = 0; i < NTILES; i++) {
      afrag16 b = load_frag(bRow0 + (long)i * 16 * K, k0, half);
      acc[i] = __builtin_amdgcn_wmma_f32_16x16x32_bf16(
          false, a, false, b, (short)0, acc[i], false, false);
    }
  }
  // C/D layout: VGPR r -> row M = r + half*8, col N = lane&15
  float* crow = C + (long)(mBase + half * 8) * NCOLS + nBase + l15;
#pragma unroll
  for (int r = 0; r < 8; r++) {
#pragma unroll
    for (int i = 0; i < NTILES; i++) {
      crow[(long)r * NCOLS + i * 16] = acc[i][r];
    }
  }
}

// Bidirectional SRU elementwise recurrence. One thread per (d, n, j).
// t is wave-uniform; prefetch U/xn rows PF steps ahead to warm L2/WGP$.
__global__ void sru_scan_kernel(const float* __restrict__ U,
                                const float* __restrict__ xn,   // null when k==4
                                const float* __restrict__ v,
                                const float* __restrict__ bvec,
                                float* __restrict__ out,
                                int hidden, int k) {
  const int PF = 16;
  int idx = blockIdx.x * 256 + threadIdx.x;   // < 2*NN*hidden
  int j  = idx % hidden;
  int nd = idx / hidden;
  int n  = nd % NN;
  int d  = nd / NN;
  int hj = d * hidden + j;
  float vf = v[hj * 2 + 0],    vr = v[hj * 2 + 1];
  float bf = bvec[hj * 2 + 0], br = bvec[hj * 2 + 1];
  const int ncols = 2 * hidden * k;
  const int D = 2 * hidden;
  float c = 0.f;
  for (int s = 0; s < TB; s++) {
    int t = d ? (TB - 1 - s) : s;
    // prefetch ahead along this thread's walk through U (and xn for k==3)
    int tp = d ? (t - PF) : (t + PF);
    if (tp >= 0 && tp < TB) {
      long pbase = (long)tp * NN + n;
      __builtin_prefetch(U + pbase * ncols + (long)hj * k, 0, 1);
      if (k == 3) __builtin_prefetch(xn + pbase * D + hj, 0, 1);
    }
    long rbase = (long)t * NN + n;
    const float* u = U + rbase * ncols + (long)hj * k;
    float xt = u[0], fr = u[1], rr = u[2];
    float res = (k == 4) ? u[3] : xn[rbase * D + hj];
    float f  = 1.f / (1.f + __expf(-(fr + vf * c + bf)));
    float c2 = f * c + (1.f - f) * xt;
    float r  = 1.f / (1.f + __expf(-(rr + vr * c + br)));
    float rl = c2 > 0.f ? c2 : 0.f;
    out[rbase * D + hj] = r * rl + (1.f - r) * res;
    c = c2;
  }
}

// Output head: out[(l*512+b)*64+e] = sum_d h[b*256+l][d]*wout[e][d] + bout[e]
__global__ void gemm_out_kernel(const unsigned short* __restrict__ A,   // ROWS x 128
                                const unsigned short* __restrict__ Wt,  // 64 x 128
                                const float* __restrict__ bias,
                                float* __restrict__ out) {
  const int K = 128;
  int wave = threadIdx.x >> 5, lane = threadIdx.x & 31;
  int half = lane >> 4, l15 = lane & 15;
  int mBase = blockIdx.x * 128 + wave * 16;
  const unsigned short* aRow = A + (long)(mBase + l15) * K;
  const unsigned short* bRow = Wt + (long)l15 * K;
  accf8 c0 = {}, c1 = {}, c2 = {}, c3 = {};
#pragma unroll
  for (int k0 = 0; k0 < K; k0 += 32) {
    afrag16 a  = load_frag(aRow, k0, half);
    afrag16 b0 = load_frag(bRow + 0L * 16 * K, k0, half);
    afrag16 b1 = load_frag(bRow + 1L * 16 * K, k0, half);
    afrag16 b2 = load_frag(bRow + 2L * 16 * K, k0, half);
    afrag16 b3 = load_frag(bRow + 3L * 16 * K, k0, half);
    c0 = __builtin_amdgcn_wmma_f32_16x16x32_bf16(false, a, false, b0, (short)0, c0, false, false);
    c1 = __builtin_amdgcn_wmma_f32_16x16x32_bf16(false, a, false, b1, (short)0, c1, false, false);
    c2 = __builtin_amdgcn_wmma_f32_16x16x32_bf16(false, a, false, b2, (short)0, c2, false, false);
    c3 = __builtin_amdgcn_wmma_f32_16x16x32_bf16(false, a, false, b3, (short)0, c3, false, false);
  }
  int col = l15;
#pragma unroll
  for (int r = 0; r < 8; r++) {
    int m  = mBase + half * 8 + r;
    int bb = m >> 8;        // m / 256 -> b
    int ll = m & 255;       // m % 256 -> l
    long o = ((long)ll * 512 + bb) * 64;
    out[o + col]      = c0[r] + bias[col];
    out[o + col + 16] = c1[r] + bias[col + 16];
    out[o + col + 32] = c2[r] + bias[col + 32];
    out[o + col + 48] = c3[r] + bias[col + 48];
  }
}

// ---------------- launch ----------------

extern "C" void kernel_launch(void* const* d_in, const int* in_sizes, int n_in,
                              void* d_out, int out_size, void* d_ws, size_t ws_size,
                              hipStream_t stream) {
  const int*   x   = (const int*)d_in[0];
  const float* emb = (const float*)d_in[1];
  const float* ng  = (const float*)d_in[2];
  const float* nb  = (const float*)d_in[3];
  const float* W[4]  = {(const float*)d_in[4],  (const float*)d_in[9],
                        (const float*)d_in[14], (const float*)d_in[19]};
  const float* Vv[4] = {(const float*)d_in[5],  (const float*)d_in[10],
                        (const float*)d_in[15], (const float*)d_in[20]};
  const float* Bb[4] = {(const float*)d_in[6],  (const float*)d_in[11],
                        (const float*)d_in[16], (const float*)d_in[21]};
  const float* Gg[4] = {(const float*)d_in[7],  (const float*)d_in[12],
                        (const float*)d_in[17], (const float*)d_in[22]};
  const float* Bt[4] = {(const float*)d_in[8],  (const float*)d_in[13],
                        (const float*)d_in[18], (const float*)d_in[23]};
  const float* wout = (const float*)d_in[24];
  const float* bout = (const float*)d_in[25];
  float* outp = (float*)d_out;

  char* ws = (char*)d_ws;
  size_t off = 0;
  auto wsalloc = [&](size_t bytes) -> void* {
    void* p = ws + off;
    off += (bytes + 255) & ~(size_t)255;
    return p;
  };
  float* buf          = (float*)wsalloc((size_t)ROWS * 256 * 4);  // activations (in-place across layers)
  float* xn           = (float*)wsalloc((size_t)ROWS * 256 * 4);  // LN'd input (f32, residual)
  unsigned short* xnb = (unsigned short*)wsalloc((size_t)ROWS * 256 * 2);  // LN'd input (bf16, GEMM A)
  float* U            = (float*)wsalloc((size_t)ROWS * 768 * 4);  // projection output
  unsigned short* wt  = (unsigned short*)wsalloc((size_t)768 * 256 * 2);   // weight, [col][k] bf16

  dim3 blk(256);

  // 1) embedding gather + outer LN
  embed_ln_kernel<<<ROWS / 8, blk, 0, stream>>>(x, emb, ng, nb, buf);

  const int Din[4] = {256, 256, 256, 128};
  const int Hh[4]  = {128, 128, 64, 64};
  const int Kk[4]  = {3, 3, 4, 3};

  for (int li = 0; li < 4; li++) {
    int D = Din[li], hidden = Hh[li], kk = Kk[li], ncols = 2 * hidden * kk;

    // weight transpose/convert
    int wn = D * ncols;
    wconv_kernel<<<(wn + 255) / 256, blk, 0, stream>>>(W[li], wt, D, ncols);

    // per-layer LN
    if (D == 256)
      ln_kernel<256><<<ROWS / 8, blk, 0, stream>>>(buf, Gg[li], Bt[li], xn, xnb);
    else
      ln_kernel<128><<<ROWS / 8, blk, 0, stream>>>(buf, Gg[li], Bt[li], xn, xnb);

    // projection GEMM (WMMA bf16 -> f32); 16x128 wave tile (8 n-tiles)
    dim3 grid(ROWS / 128, ncols / 128);
    if (li <= 1)      gemm_bf16_kernel<256, 768, 8><<<grid, blk, 0, stream>>>(xnb, wt, U);
    else if (li == 2) gemm_bf16_kernel<256, 512, 8><<<grid, blk, 0, stream>>>(xnb, wt, U);
    else              gemm_bf16_kernel<128, 384, 8><<<grid, blk, 0, stream>>>(xnb, wt, U);

    // bidirectional recurrence; writes back into `buf` (input is dead post-LN)
    int threads = 2 * NN * hidden;
    sru_scan_kernel<<<threads / 256, blk, 0, stream>>>(
        U, (kk == 3) ? xn : nullptr, Vv[li], Bb[li], buf, hidden, kk);
  }

  // 3) output head: wout is already [e][d] == Wt layout; convert + GEMM + remap
  f2b_kernel<<<(64 * 128 + 255) / 256, blk, 0, stream>>>(wout, wt, 64 * 128);
  long nA = (long)ROWS * 128;
  f2b_kernel<<<(int)((nA + 255) / 256), blk, 0, stream>>>(buf, xnb, nA);
  gemm_out_kernel<<<ROWS / 128, blk, 0, stream>>>(xnb, wt, bout, outp);
}
